// RecurrentGeneralizedPFTransformer_85246510891773
// MI455X (gfx1250) — matrix operand order repaired and checked by reading, not compiled, not run on hardware
//
#include <hip/hip_runtime.h>
#include <hip/hip_bf16.h>
#include <math.h>

typedef __bf16 bf16;
typedef __bf16 v8bf  __attribute__((ext_vector_type(8)));
typedef __bf16 v16bf __attribute__((ext_vector_type(16)));
typedef float  v8f   __attribute__((ext_vector_type(8)));

#define B_    2
#define N_    1024
#define D_    512
#define H_    8
#define DH_   64
#define HD_   512
#define FF_   2048
#define DEPTH_ 4

// ---------------------------------------------------------------------------
// CDNA5 async global->LDS copy (16B per lane), ASYNCcnt-tracked.
// ---------------------------------------------------------------------------
__device__ __forceinline__ void async_cp16(unsigned int lds_addr, const bf16* gptr) {
  asm volatile("global_load_async_to_lds_b128 %0, %1, off"
               :: "v"(lds_addr), "v"((unsigned long long)(uintptr_t)gptr)
               : "memory");
}
__device__ __forceinline__ void async_wait0() {
  asm volatile("s_wait_asynccnt 0x0" ::: "memory");
}

// ---------------------------------------------------------------------------
// utility kernels
// ---------------------------------------------------------------------------
__global__ void k_f32_to_bf16(const float* __restrict__ s, bf16* __restrict__ d, long n) {
  long i = (long)blockIdx.x * blockDim.x + threadIdx.x;
  long st = (long)gridDim.x * blockDim.x;
  for (; i < n; i += st) d[i] = (bf16)s[i];
}

__global__ void k_copy_f32(const float* __restrict__ s, float* __restrict__ d, long n) {
  long i = (long)blockIdx.x * blockDim.x + threadIdx.x;
  long st = (long)gridDim.x * blockDim.x;
  for (; i < n; i += st) d[i] = s[i];
}

// Fused transpose + f32->bf16: src (L,R,C) -> dst (L,C,R). Tiled 32x32 via LDS.
__global__ void k_transpose_bf16(const float* __restrict__ src, bf16* __restrict__ dst,
                                 int R, int C) {
  __shared__ float tile[32][33];
  const int l  = blockIdx.z;
  const int r0 = blockIdx.y * 32, c0 = blockIdx.x * 32;
  const float* s = src + (long)l * R * C;
  bf16* d = dst + (long)l * R * C;
  const int tx = threadIdx.x, ty = threadIdx.y;     // 32 x 8
  #pragma unroll
  for (int i = ty; i < 32; i += 8)
    tile[i][tx] = s[(long)(r0 + i) * C + c0 + tx];
  __syncthreads();
  #pragma unroll
  for (int i = ty; i < 32; i += 8)
    d[(long)(c0 + i) * R + r0 + tx] = (bf16)tile[tx][i];
}

// LayerNorm over last dim (D) of (rows, D), emit bf16
__global__ void k_layernorm_bf16(const float* __restrict__ x, const float* __restrict__ g,
                                 const float* __restrict__ b, bf16* __restrict__ out, int D) {
  __shared__ float s1[256];
  __shared__ float s2[256];
  const int row = blockIdx.x;
  const int tid = threadIdx.x;
  const float* xr = x + (long)row * D;
  float a = 0.f, q = 0.f;
  for (int i = tid; i < D; i += 256) { float v = xr[i]; a += v; q += v * v; }
  s1[tid] = a; s2[tid] = q; __syncthreads();
  for (int o = 128; o > 0; o >>= 1) {
    if (tid < o) { s1[tid] += s1[tid + o]; s2[tid] += s2[tid + o]; }
    __syncthreads();
  }
  const float mean = s1[0] / D;
  const float var  = s2[0] / D - mean * mean;
  const float inv  = rsqrtf(var + 1e-5f);
  for (int i = tid; i < D; i += 256)
    out[(long)row * D + i] = (bf16)((xr[i] - mean) * inv * g[i] + b[i]);
}

// qkv (b,n,3*HD) f32 -> Q/K (b,h,n,d) bf16 (+bias_pf on Q), V transposed (b,h,d,n)
__global__ void k_pack_qkv(const float* __restrict__ qkv, const float* __restrict__ bias_pf,
                           bf16* __restrict__ Q, bf16* __restrict__ Kd, bf16* __restrict__ Vt) {
  long idx = (long)blockIdx.x * blockDim.x + threadIdx.x;   // ((b*H+h)*N+n)*DH+d
  if (idx >= (long)B_ * H_ * N_ * DH_) return;
  int d = (int)(idx % DH_); long t = idx / DH_;
  int n = (int)(t % N_);    t /= N_;
  int h = (int)(t % H_);    int b = (int)(t / H_);
  long src = ((long)b * N_ + n) * (3 * HD_) + h * DH_ + d;
  Q[idx]  = (bf16)(qkv[src] + bias_pf[h * DH_ + d]);
  Kd[idx] = (bf16)qkv[src + HD_];
  Vt[(((long)b * H_ + h) * DH_ + d) * N_ + n] = (bf16)qkv[src + 2 * HD_];
}

// (b,n,HD) f32 -> (b,h,n,d) bf16
__global__ void k_pack_heads(const float* __restrict__ src, bf16* __restrict__ dst) {
  long idx = (long)blockIdx.x * blockDim.x + threadIdx.x;
  if (idx >= (long)B_ * H_ * N_ * DH_) return;
  int d = (int)(idx % DH_); long t = idx / DH_;
  int n = (int)(t % N_);    t /= N_;
  int h = (int)(t % H_);    int b = (int)(t / H_);
  dst[idx] = (bf16)src[((long)b * N_ + n) * HD_ + h * DH_ + d];
}

// (b,h,n,d) f32 -> (b,n,h*DH+d) bf16
__global__ void k_unpack_heads(const float* __restrict__ src, bf16* __restrict__ dst) {
  long idx = (long)blockIdx.x * blockDim.x + threadIdx.x;
  if (idx >= (long)B_ * H_ * N_ * DH_) return;
  int d = (int)(idx % DH_); long t = idx / DH_;
  int n = (int)(t % N_);    t /= N_;
  int h = (int)(t % H_);    int b = (int)(t / H_);
  dst[((long)b * N_ + n) * HD_ + h * DH_ + d] = (bf16)src[idx];
}

// Fused Transformer-XL rel_shift + (AC+BD)*scale + softmax over HEADS.
// AC/BD layout (b,h,i,j); attn written in place into AC + bf16 copy.
__global__ void k_relshift_softmax(float* __restrict__ AC, const float* __restrict__ BD,
                                   bf16* __restrict__ attn_bf) {
  long idx = (long)blockIdx.x * blockDim.x + threadIdx.x;   // (b*N+i)*N+j
  if (idx >= (long)B_ * N_ * N_) return;
  int j = (int)(idx % N_); long t = idx / N_;
  int i = (int)(t % N_);   int b = (int)(t / N_);
  int i2 = 0, j2 = 0; bool z = false;
  if (j <= i)           { i2 = i;     j2 = j - i + N_ - 1; }
  else if (j == i + 1)  { z = true; }
  else                  { i2 = i + 1; j2 = j - i - 2; }
  float vals[H_];
  float mx = -3.0e38f;
  #pragma unroll
  for (int h = 0; h < H_; ++h) {
    long base = (long)(b * H_ + h) * N_;
    float ac = AC[(base + i) * N_ + j];
    float bd = z ? 0.f : BD[(base + i2) * N_ + j2];
    float v = (ac + bd) * 0.125f;   // SCALE = DH^-0.5 ; BD carries 1/3 via Wsum
    vals[h] = v; mx = fmaxf(mx, v);
  }
  float sum = 0.f;
  #pragma unroll
  for (int h = 0; h < H_; ++h) { vals[h] = expf(vals[h] - mx); sum += vals[h]; }
  const float inv = 1.f / sum;
  #pragma unroll
  for (int h = 0; h < H_; ++h) {
    long off = ((long)(b * H_ + h) * N_ + i) * N_ + j;
    float a = vals[h] * inv;
    AC[off] = a;
    attn_bf[off] = (bf16)a;
  }
}

// ---------------------------------------------------------------------------
// WMMA bf16 GEMM: C[bz] = alpha * A[bz](MxK) * BT[bz](NxK)^T [+bias +res] [gelu]
// B operand ALWAYS supplied transposed (N x K row-major) -> contiguous staging.
// Async global->LDS staging with double-buffered LDS; 8 waves/block.
// ---------------------------------------------------------------------------
template<int BM, int BN, int BK, int WM, int WN>
__global__ __launch_bounds__(256)
void k_gemm_bf16_wmma(const bf16* __restrict__ A, const bf16* __restrict__ BT,
                      float* __restrict__ C, bf16* __restrict__ Cbf,
                      const float* __restrict__ bias, const float* __restrict__ res,
                      float alpha, int K, int lda, int ldb, int ldc,
                      long sA, long sB, long sC, int gelu) {
  constexpr int WAVES_M = BM / WM;
  constexpr int WAVES_N = BN / WN;
  static_assert(WAVES_M * WAVES_N == 8, "need 8 waves");
  constexpr int FM = WM / 16;
  constexpr int FN = WN / 16;
  constexpr int ACP = (BM * BK) / (256 * 8);
  constexpr int BCP = (BN * BK) / (256 * 8);
  static_assert(ACP >= 1 && BCP >= 1, "tile too small");

  __shared__ alignas(16) bf16 As[2][BM * BK];   // [m][k]
  __shared__ alignas(16) bf16 Bs[2][BN * BK];   // [n][k]

  const int tid  = threadIdx.x;
  const int lane = tid & 31;
  const int wv   = tid >> 5;
  const int bn0  = blockIdx.x * BN;
  const int bm0  = blockIdx.y * BM;
  const int bz   = blockIdx.z;
  const bf16* Ab = A + (long)bz * sA;
  const bf16* Bb = BT + (long)bz * sB;
  const long cbase = (long)bz * sC;
  const int wm = (wv % WAVES_M) * WM;
  const int wn = (wv / WAVES_M) * WN;

  const unsigned int aLds[2] = { (unsigned int)(uintptr_t)&As[0][0],
                                 (unsigned int)(uintptr_t)&As[1][0] };
  const unsigned int bLds[2] = { (unsigned int)(uintptr_t)&Bs[0][0],
                                 (unsigned int)(uintptr_t)&Bs[1][0] };

  v8f acc[FM][FN];
  #pragma unroll
  for (int i = 0; i < FM; ++i)
    #pragma unroll
    for (int j = 0; j < FN; ++j)
      #pragma unroll
      for (int e = 0; e < 8; ++e) acc[i][j][e] = 0.0f;

  const int half = lane >> 4;       // lane half selects K sub-range per ISA layout
  const int l16  = lane & 15;
  const int kaA  = half ? 8 : 0;    // A: lanes16-31 hold K=8..15 / 24..31
  const int kaB  = half ? 8 : 0;    // B: lanes16-31 hold K=8..15 / 24..31

  auto stage = [&](int k0, int p) {
    #pragma unroll
    for (int it = 0; it < ACP; ++it) {
      int e = (it * 256 + tid) * 8;
      int r = e / BK, c = e % BK;
      async_cp16(aLds[p] + (unsigned int)(r * BK + c) * 2,
                 Ab + (long)(bm0 + r) * lda + k0 + c);
    }
    #pragma unroll
    for (int it = 0; it < BCP; ++it) {
      int e = (it * 256 + tid) * 8;
      int r = e / BK, c = e % BK;
      async_cp16(bLds[p] + (unsigned int)(r * BK + c) * 2,
                 Bb + (long)(bn0 + r) * ldb + k0 + c);
    }
  };

  stage(0, 0);
  async_wait0();
  __syncthreads();

  int p = 0;
  for (int k0 = 0; k0 < K; k0 += BK) {
    if (k0 + BK < K) stage(k0 + BK, p ^ 1);   // overlap next slab with WMMAs

    v16bf afr[FM];
    v16bf bfr[FN];
    #pragma unroll
    for (int i = 0; i < FM; ++i) {
      const bf16* pa = &As[p][(wm + i * 16 + l16) * BK];
      v8bf lo = *(const v8bf*)(pa + kaA);
      v8bf hi = *(const v8bf*)(pa + kaA + 16);
      afr[i] = __builtin_shufflevector(lo, hi, 0,1,2,3,4,5,6,7,8,9,10,11,12,13,14,15);
    }
    #pragma unroll
    for (int j = 0; j < FN; ++j) {
      const bf16* pb = &Bs[p][(wn + j * 16 + l16) * BK];
      v8bf lo = *(const v8bf*)(pb + kaB);
      v8bf hi = *(const v8bf*)(pb + kaB + 16);
      bfr[j] = __builtin_shufflevector(lo, hi, 0,1,2,3,4,5,6,7,8,9,10,11,12,13,14,15);
    }
    #pragma unroll
    for (int i = 0; i < FM; ++i)
      #pragma unroll
      for (int j = 0; j < FN; ++j)
        acc[i][j] = __builtin_amdgcn_wmma_f32_16x16x32_bf16(
            false, afr[i], false, bfr[j], (short)0, acc[i][j], false, false);

    async_wait0();        // next slab landed
    __syncthreads();
    p ^= 1;
  }

  // ---- epilogue: C/D layout (VGPR e -> M = e (+8 by lane half); N = lane&15) ----
  const int mo = half * 8;
  #pragma unroll
  for (int i = 0; i < FM; ++i) {
    #pragma unroll
    for (int j = 0; j < FN; ++j) {
      #pragma unroll
      for (int e = 0; e < 8; ++e) {
        int gm = bm0 + wm + i * 16 + mo + e;
        int gn = bn0 + wn + j * 16 + l16;
        long ci = cbase + (long)gm * ldc + gn;
        float v = acc[i][j][e] * alpha;
        if (bias) v += bias[gn];
        if (res)  v += res[ci];
        if (gelu) v = 0.5f * v * (1.0f + erff(v * 0.70710678118f));
        if (C)   C[ci] = v;
        if (Cbf) Cbf[ci] = (bf16)v;
      }
    }
  }
}

// ---------------------------------------------------------------------------
extern "C" void kernel_launch(void* const* d_in, const int* in_sizes, int n_in,
                              void* d_out, int out_size, void* d_ws, size_t ws_size,
                              hipStream_t stream) {
  (void)in_sizes; (void)n_in; (void)out_size; (void)ws_size;
  const float* in_x    = (const float*)d_in[0];
  const float* in_rt   = (const float*)d_in[1];
  const float* in_rc   = (const float*)d_in[2];
  const float* in_rp   = (const float*)d_in[3];
  const float* bias_pf = (const float*)d_in[4];
  const float* ln1_g   = (const float*)d_in[5];
  const float* ln1_b   = (const float*)d_in[6];
  const float* w_qkv   = (const float*)d_in[7];
  const float* w_out   = (const float*)d_in[8];
  const float* b_out   = (const float*)d_in[9];
  const float* w_kt    = (const float*)d_in[10];
  const float* w_kc    = (const float*)d_in[11];
  const float* w_kp    = (const float*)d_in[12];
  const float* ln2_g   = (const float*)d_in[13];
  const float* ln2_b   = (const float*)d_in[14];
  const float* w_ff1   = (const float*)d_in[15];
  const float* b_ff1   = (const float*)d_in[16];
  const float* w_ff2   = (const float*)d_in[17];
  const float* b_ff2   = (const float*)d_in[18];
  float* out = (float*)d_out;

  // ---- workspace bump allocator ----
  char* ws = (char*)d_ws;
  size_t off = 0;
  auto alloc = [&](size_t bytes) -> void* {
    void* p = ws + off;
    off = (off + bytes + 255) & ~(size_t)255;
    return p;
  };
  const long NWQ  = (long)DEPTH_ * D_ * 3 * HD_;
  const long NWO  = (long)DEPTH_ * HD_ * D_;
  const long NWK  = (long)DEPTH_ * D_ * HD_;
  const long NWF1 = (long)DEPTH_ * D_ * FF_;
  const long NWF2 = (long)DEPTH_ * FF_ * D_;
  const long NR   = (long)B_ * N_ * D_;
  const long NBH  = (long)B_ * H_ * N_ * DH_;
  const long NSC  = (long)B_ * H_ * N_ * N_;

  // all weight copies stored TRANSPOSED (per layer: (Ncols, K)) as bf16
  bf16* wq_bf  = (bf16*)alloc(NWQ  * 2);
  bf16* wo_bf  = (bf16*)alloc(NWO  * 2);
  bf16* wkt_bf = (bf16*)alloc(NWK  * 2);
  bf16* wkc_bf = (bf16*)alloc(NWK  * 2);
  bf16* wkp_bf = (bf16*)alloc(NWK  * 2);
  bf16* wf1_bf = (bf16*)alloc(NWF1 * 2);
  bf16* wf2_bf = (bf16*)alloc(NWF2 * 2);
  bf16* rt_bf  = (bf16*)alloc(NR * 2);
  bf16* rc_bf  = (bf16*)alloc(NR * 2);
  bf16* rp_bf  = (bf16*)alloc(NR * 2);
  bf16* xn_bf  = (bf16*)alloc(NR * 2);
  float* qkv_f = (float*)alloc((long)B_ * N_ * 3 * HD_ * 4);
  bf16* Qbf    = (bf16*)alloc(NBH * 2);
  bf16* Kbf    = (bf16*)alloc(NBH * 2);
  bf16* Vtbf   = (bf16*)alloc(NBH * 2);       // (b,h,d,n)
  float* wsum_f = (float*)alloc(NR * 4);
  bf16* ws_bf  = (bf16*)alloc(NBH * 2);
  float* AC_f  = (float*)alloc(NSC * 4);
  float* BD_f  = (float*)alloc(NSC * 4);
  bf16* attn_bf = (bf16*)alloc(NSC * 2);
  float* ao_f  = (float*)alloc(NBH * 4);
  bf16* ho_bf  = (bf16*)alloc(NBH * 2);
  float* xbuf  = (float*)alloc(NR * 4);
  bf16* h1_bf  = (bf16*)alloc((long)B_ * N_ * FF_ * 2);

  auto cvt = [&](const float* s, bf16* d, long n) {
    k_f32_to_bf16<<<1024, 256, 0, stream>>>(s, d, n);
  };
  auto cvtT = [&](const float* s, bf16* d, int R, int Cc) {
    dim3 g(Cc / 32, R / 32, DEPTH_), b(32, 8);
    k_transpose_bf16<<<g, b, 0, stream>>>(s, d, R, Cc);
  };
  cvtT(w_qkv, wq_bf, D_, 3 * HD_);
  cvtT(w_out, wo_bf, HD_, D_);
  cvtT(w_kt, wkt_bf, D_, HD_);
  cvtT(w_kc, wkc_bf, D_, HD_);
  cvtT(w_kp, wkp_bf, D_, HD_);
  cvtT(w_ff1, wf1_bf, D_, FF_);
  cvtT(w_ff2, wf2_bf, FF_, D_);
  cvt(in_rt, rt_bf, NR);  cvt(in_rc, rc_bf, NR);  cvt(in_rp, rp_bf, NR);
  k_copy_f32<<<1024, 256, 0, stream>>>(in_x, xbuf, NR);

  const int M = B_ * N_;            // 2048 rows
  const dim3 blk(256);
  const int gPack = (int)((NBH + 255) / 256);
  const int gSm   = (int)(((long)B_ * N_ * N_ + 255) / 256);

  for (int l = 0; l < DEPTH_; ++l) {
    // ---- LN1 -> bf16 ----
    k_layernorm_bf16<<<M, blk, 0, stream>>>(xbuf, ln1_g + l * D_, ln1_b + l * D_, xn_bf, D_);

    // ---- qkv = xn @ w_qkv[l]  (BT = (1536, 512)) ----
    {
      dim3 g(3 * HD_ / 128, M / 128, 1);
      k_gemm_bf16_wmma<128,128,32,32,64><<<g, blk, 0, stream>>>(
          xn_bf, wq_bf + (long)l * D_ * 3 * HD_, qkv_f, nullptr, nullptr, nullptr,
          1.0f, D_, D_, D_, 3 * HD_, 0, 0, 0, 0);
    }
    k_pack_qkv<<<gPack, blk, 0, stream>>>(qkv_f, bias_pf, Qbf, Kbf, Vtbf);

    // ---- Wsum = (r_t@w_kt + r_c@w_kc + r_p@w_kp)/3  (rel_shift is linear) ----
    {
      dim3 g(HD_ / 128, M / 128, 1);
      k_gemm_bf16_wmma<128,128,32,32,64><<<g, blk, 0, stream>>>(
          rt_bf, wkt_bf + (long)l * D_ * HD_, wsum_f, nullptr, nullptr, nullptr,
          1.0f / 3.0f, D_, D_, D_, HD_, 0, 0, 0, 0);
      k_gemm_bf16_wmma<128,128,32,32,64><<<g, blk, 0, stream>>>(
          rc_bf, wkc_bf + (long)l * D_ * HD_, wsum_f, nullptr, nullptr, wsum_f,
          1.0f / 3.0f, D_, D_, D_, HD_, 0, 0, 0, 0);
      k_gemm_bf16_wmma<128,128,32,32,64><<<g, blk, 0, stream>>>(
          rp_bf, wkp_bf + (long)l * D_ * HD_, wsum_f, nullptr, nullptr, wsum_f,
          1.0f / 3.0f, D_, D_, D_, HD_, 0, 0, 0, 0);
    }
    k_pack_heads<<<gPack, blk, 0, stream>>>(wsum_f, ws_bf);

    // ---- scores: AC = Q K^T, BDraw = Q Wsum^T  (batched over b*h = 16) ----
    {
      dim3 g(N_ / 128, N_ / 128, B_ * H_);
      k_gemm_bf16_wmma<128,128,32,32,64><<<g, blk, 0, stream>>>(
          Qbf, Kbf, AC_f, nullptr, nullptr, nullptr,
          1.0f, DH_, DH_, DH_, N_, (long)N_ * DH_, (long)N_ * DH_, (long)N_ * N_, 0);
      k_gemm_bf16_wmma<128,128,32,32,64><<<g, blk, 0, stream>>>(
          Qbf, ws_bf, BD_f, nullptr, nullptr, nullptr,
          1.0f, DH_, DH_, DH_, N_, (long)N_ * DH_, (long)N_ * DH_, (long)N_ * N_, 0);
    }

    // ---- rel_shift + scale + softmax over heads ----
    k_relshift_softmax<<<gSm, blk, 0, stream>>>(AC_f, BD_f, attn_bf);

    // ---- out = attn @ V (Vt is (b,h,d,n)) ----
    {
      dim3 g(DH_ / 64, N_ / 128, B_ * H_);
      k_gemm_bf16_wmma<128,64,32,32,32><<<g, blk, 0, stream>>>(
          attn_bf, Vtbf, ao_f, nullptr, nullptr, nullptr,
          1.0f, N_, N_, N_, DH_, (long)N_ * N_, (long)N_ * DH_, (long)N_ * DH_, 0);
    }
    k_unpack_heads<<<gPack, blk, 0, stream>>>(ao_f, ho_bf);

    // ---- x = ho @ w_out[l] + b_out[l] + x  (BT = (D, HD)) ----
    {
      dim3 g(D_ / 128, M / 128, 1);
      k_gemm_bf16_wmma<128,128,32,32,64><<<g, blk, 0, stream>>>(
          ho_bf, wo_bf + (long)l * HD_ * D_, xbuf, nullptr, b_out + l * D_, xbuf,
          1.0f, HD_, HD_, HD_, D_, 0, 0, 0, 0);
    }

    // ---- LN2 -> bf16 ----
    k_layernorm_bf16<<<M, blk, 0, stream>>>(xbuf, ln2_g + l * D_, ln2_b + l * D_, xn_bf, D_);

    // ---- h1 = gelu(xn2 @ w_ff1 + b_ff1) -> bf16 (BT = (FF, D)) ----
    {
      dim3 g(FF_ / 128, M / 128, 1);
      k_gemm_bf16_wmma<128,128,32,32,64><<<g, blk, 0, stream>>>(
          xn_bf, wf1_bf + (long)l * D_ * FF_, nullptr, h1_bf, b_ff1 + l * FF_, nullptr,
          1.0f, D_, D_, D_, FF_, 0, 0, 0, 1);
    }
    // ---- x = h1 @ w_ff2 + b_ff2 + x  (BT = (D, FF)) ----
    {
      dim3 g(D_ / 128, M / 128, 1);
      k_gemm_bf16_wmma<128,128,32,32,64><<<g, blk, 0, stream>>>(
          h1_bf, wf2_bf + (long)l * FF_ * D_, xbuf, nullptr, b_ff2 + l * D_, xbuf,
          1.0f, FF_, FF_, FF_, D_, 0, 0, 0, 0);
    }
  }

  // ---- outputs: x (B,N,D) then attn_last already in (b,h,q,k) layout ----
  k_copy_f32<<<1024, 256, 0, stream>>>(xbuf, out, NR);
  k_copy_f32<<<4096, 256, 0, stream>>>(AC_f, out + NR, NSC);
}
